// ProjectionScalePolar_28630251995466
// MI455X (gfx1250) — matrix operand order repaired and checked by reading, not compiled, not run on hardware
//
#include <hip/hip_runtime.h>

typedef __attribute__((ext_vector_type(16))) _Float16 v16h;
typedef __attribute__((ext_vector_type(8)))  _Float16 v8h;
typedef __attribute__((ext_vector_type(8)))  float    v8f;

#define B_   8
#define C_   128
#define H_   128
#define W_   352
#define S_   33
#define D_   64
#define WT   2           // w values per block (vectorizes loads/stores)
#define KCH  32          // K chunk staged per GEMM step
#define HPAD 144         // padded K length (halves) for B panels: 288B rows, 32B aligned
#define APAD (KCH + 8)   // padded row (halves) for A chunks: 80B rows, 16B aligned
#define PSW  (S_ + 3)    // padded pixel_scales row

// phase-disjoint union region: pixel_scales panel (softmax) vs A staging (GEMM)
#define DYN_BYTES_PS (H_ * PSW * 4)               // 18432
#define DYN_BYTES_A  (WT * C_ * APAD * 2)         // 20480
#define DYN_BYTES    (DYN_BYTES_A > DYN_BYTES_PS ? DYN_BYTES_A : DYN_BYTES_PS)

union V16H { v16h v; v8h h[2]; };

// ---- gfx1250 async global->LDS staging (guarded; falls back to sync path) ----
#if __has_builtin(__builtin_amdgcn_global_load_async_to_lds_b32)
#define HAVE_ASYNC_LDS 1
typedef __attribute__((address_space(1))) int* gptr_t;
typedef __attribute__((address_space(3))) int* lptr_t;
// generic->AS1: full 64-bit value is the global address
__device__ __forceinline__ gptr_t as_global(const void* p) {
    return (gptr_t)(unsigned long long)(uintptr_t)p;
}
// generic->AS3: low 32 bits of a flat LDS pointer are the wave-relative LDS offset
__device__ __forceinline__ lptr_t as_local(const void* p) {
    return (lptr_t)(unsigned int)(uintptr_t)p;
}
__device__ __forceinline__ void wait_async0() {
#if __has_builtin(__builtin_amdgcn_s_wait_asynccnt)
    __builtin_amdgcn_s_wait_asynccnt(0);
#else
    asm volatile("s_wait_asynccnt 0x0" ::: "memory");
#endif
}
#endif

__global__ __launch_bounds__(256)
void psp_fused_kernel(const float* __restrict__ image,
                      const float* __restrict__ pixel_scales,
                      const float* __restrict__ fvec,
                      float* __restrict__ out)
{
    const int w0  = blockIdx.x * WT;
    const int b   = blockIdx.y;
    const int tid = threadIdx.x;

    __shared__ __align__(32) _Float16      s_prob[WT * D_ * HPAD]; // B panels [wi][d][h]
    __shared__ __align__(32) unsigned char s_dyn[DYN_BYTES];       // ps panel / A chunks
    __shared__ int   s_i0[D_], s_i1[D_];
    __shared__ float s_w0[D_], s_w1[D_];

    float (*s_ps)[PSW] = reinterpret_cast<float (*)[PSW]>(s_dyn);  // [h][s]
    _Float16* s_a      = reinterpret_cast<_Float16*>(s_dyn);       // [wi][c][k]

    // ---- per-depth-bin interpolation params: depend only on (b, d) ----
    if (tid < D_) {
        const float step_flip = 32.0f - 0.5f * (float)tid;          // steps reversed
        const float sc        = fvec[b * 2 + 1] / step_flip;
        const float log_norm  = __log2f(sc) / 9.0f * 2.0f - 1.0f;   // SCALE_[MIN,MAX]=[0,9]
        const float pos       = (log_norm + 1.0f) * 0.5f * (float)(S_ - 1);
        const float i0f       = floorf(pos);
        const float w1        = pos - i0f;
        int i0 = (int)i0f;
        int i1 = i0 + 1;
        const float ok0 = (i0 >= 0 && i0 <= S_ - 1) ? 1.0f : 0.0f;
        const float ok1 = (i1 >= 0 && i1 <= S_ - 1) ? 1.0f : 0.0f;
        i0 = min(max(i0, 0), S_ - 1);
        i1 = min(max(i1, 0), S_ - 1);
        s_i0[tid] = i0;  s_i1[tid] = i1;
        s_w0[tid] = (1.0f - w1) * ok0;
        s_w1[tid] = w1 * ok1;
    }

    // ---- softmax panels for both w values ----
    for (int wi = 0; wi < WT; ++wi) {
        __syncthreads();   // previous user of s_dyn done (also covers interp params)

        // stage pixel_scales rows for (b, w0+wi): 128 x 33 f32, raw bytes -> LDS
        for (int idx = tid; idx < H_ * S_; idx += 256) {
            const int h = idx / S_;
            const int s = idx - h * S_;
            const float* g = &pixel_scales[(((size_t)b * H_ + h) * W_ + (w0 + wi)) * S_ + s];
#ifdef HAVE_ASYNC_LDS
            __builtin_amdgcn_global_load_async_to_lds_b32(
                as_global(g), as_local(&s_ps[h][s]), /*offset=*/0, /*cpol=*/0);
#else
            s_ps[h][s] = *g;
#endif
        }
#ifdef HAVE_ASYNC_LDS
        wait_async0();     // this wave's async copies landed in LDS
#endif
        __syncthreads();

        if (tid < H_) {   // one thread per h row
            const int h = tid;
            float mx = -3.4e38f;
            for (int d = 0; d < D_; ++d) {
                const float s = s_ps[h][s_i0[d]] * s_w0[d] + s_ps[h][s_i1[d]] * s_w1[d];
                mx = fmaxf(mx, s);
            }
            float sum = 0.0f;
            for (int d = 0; d < D_; ++d) {
                const float s = s_ps[h][s_i0[d]] * s_w0[d] + s_ps[h][s_i1[d]] * s_w1[d];
                sum += __expf(s - mx);
            }
            const float inv = 1.0f / sum;
            float dabs = 0.0f;
            for (int d = 0; d < D_; ++d) {
                const float s = s_ps[h][s_i0[d]] * s_w0[d] + s_ps[h][s_i1[d]] * s_w1[d];
                const float p = __expf(s - mx) * inv;
                s_prob[(wi * D_ + d) * HPAD + h] = (_Float16)p;   // B layout: [d][h]
                dabs += p * (32.0f - 0.5f * (float)d);            // E[steps_flip]
            }
            out[(size_t)B_ * C_ * D_ * W_ + ((size_t)b * H_ + h) * W_ + (w0 + wi)] = dabs;
        }
    }
    __syncthreads();   // softmax reads of s_ps done before A staging reuses s_dyn

    // ---- dual-w GEMM: polar[c,d,w0+wi] = sum_h image[c,h,w0+wi] * prob[wi][h,d] ----
    const int wave = tid >> 5;
    const int lane = tid & 31;
    const int m16  = lane & 15;   // A row / B column within tile
    const int hi   = lane >> 4;   // K-half select
    const int c0   = wave * 16;   // this wave's M strip

    v8f acc[WT][4];
#pragma unroll
    for (int wi = 0; wi < WT; ++wi)
#pragma unroll
        for (int t = 0; t < 4; ++t) acc[wi][t] = (v8f){};

    const float* imgB = image + ((size_t)b * C_) * H_ * W_ + w0;

    for (int k0 = 0; k0 < H_; k0 += KCH) {
        // stage A K-chunk for both w: 8B vector loads along w, f16 convert en route
        for (int idx = tid; idx < C_ * KCH; idx += 256) {
            const int c = idx >> 5;
            const int k = idx & (KCH - 1);
            const float2 v = *(const float2*)&imgB[((size_t)c * H_ + (k0 + k)) * W_];
            s_a[(0 * C_ + c) * APAD + k] = (_Float16)v.x;
            s_a[(1 * C_ + c) * APAD + k] = (_Float16)v.y;
        }
        if (k0 + KCH < H_) {  // prefetch next image K-chunk into cache
            __builtin_prefetch(&imgB[((size_t)(tid >> 1) * H_ + (k0 + KCH + (tid & 1) * 16)) * W_], 0, 0);
        }
        __syncthreads();

        // A fragments: lane m16 = row; K = hi*16..+15 contiguous
        V16H af[WT];
#pragma unroll
        for (int wi = 0; wi < WT; ++wi) {
            af[wi].h[0] = *(const v8h*)&s_a[(wi * C_ + c0 + m16) * APAD + hi * 16];
            af[wi].h[1] = *(const v8h*)&s_a[(wi * C_ + c0 + m16) * APAD + hi * 16 + 8];
        }

        // B fragments: lane m16 = column d; lanes 0-15 K=k0..+15, lanes 16-31 K=k0+16..+31
#pragma unroll
        for (int wi = 0; wi < WT; ++wi) {
#pragma unroll
            for (int t = 0; t < 4; ++t) {
                const v16h bf = *(const v16h*)
                    &s_prob[(wi * D_ + t * 16 + m16) * HPAD + k0 + hi * 16];
                acc[wi][t] = __builtin_amdgcn_wmma_f32_16x16x32_f16(
                    false, af[wi].v, false, bf, (short)0, acc[wi][t], false, false);
            }
        }
        __syncthreads();   // protect s_a before next staging pass
    }

    // ---- store D tiles: VGPR r -> M = r + 8*hi, lane m16 -> N; float2 along w ----
    float* outB = out + (((size_t)b * C_) * D_) * W_ + w0;
#pragma unroll
    for (int t = 0; t < 4; ++t) {
#pragma unroll
        for (int r = 0; r < 8; ++r) {
            const int c = c0 + r + 8 * hi;
            const int d = t * 16 + m16;
            const float2 v = { acc[0][t][r], acc[1][t][r] };
            *(float2*)&outB[((size_t)c * D_ + d) * W_] = v;
        }
    }
}

extern "C" void kernel_launch(void* const* d_in, const int* in_sizes, int n_in,
                              void* d_out, int out_size, void* d_ws, size_t ws_size,
                              hipStream_t stream) {
    (void)in_sizes; (void)n_in; (void)out_size; (void)d_ws; (void)ws_size;
    const float* image        = (const float*)d_in[0];
    // d_in[1] = valid_image (unused by the reference math)
    const float* pixel_scales = (const float*)d_in[2];
    const float* fvec         = (const float*)d_in[3];
    float* out = (float*)d_out;

    dim3 grid(W_ / WT, B_);
    psp_fused_kernel<<<grid, 256, 0, stream>>>(image, pixel_scales, fvec, out);
}
